// DepthNet_78486232367646
// MI455X (gfx1250) — matrix operand order repaired and checked by reading, not compiled, not run on hardware
//
#include <hip/hip_runtime.h>
#include <cstdint>
#include <cstddef>

// ---------------------------------------------------------------------------
// DepthNet for MI455X (gfx1250): implicit-GEMM convs on v_wmma_f32_16x16x32_bf16,
// NHWC bf16 activations (L2-resident), async global->LDS weight staging
// (ASYNCcnt double buffer), deformable conv with WMMA grouped GEMM.
// ---------------------------------------------------------------------------

typedef __attribute__((ext_vector_type(16))) __bf16 v16bf;
typedef __attribute__((ext_vector_type(8)))  __bf16 v8bf;
typedef __attribute__((ext_vector_type(8)))  float  v8f;

#define CAT16(lo, hi) __builtin_shufflevector(lo, hi, 0,1,2,3,4,5,6,7,8,9,10,11,12,13,14,15)

namespace cfg {
constexpr int B = 8, H = 32, W = 88;
constexpr int HW = H * W;            // 2816
constexpr int P  = B * HW;           // 22528 spatial positions
constexpr int MTILES = P / 16;       // 1408 (exact)
constexpr int MBLK   = MTILES / 8;   // 176 blocks.x, 8 waves / block
}

// ======================= generic WMMA implicit-GEMM conv ====================
// One wave: 16 spatial positions x (NT*16) output channels.
// 8 waves per block share one output-channel group -> weight chunk staged in
// LDS via async loads, double buffered.
// OUTMODE: 0 = bf16 NHWC (stride/coff), 1 = f32 NHWC (offset map), 2 = f32 NCHW clamped.
template<int NT, int OUTMODE, bool RELU, bool HASRES>
__global__ __launch_bounds__(256)
void conv_wmma(const __bf16* __restrict__ in, const __bf16* __restrict__ wp,
               const float* __restrict__ bias, const __bf16* res,
               void* outv, int Cin, int CoutP, int khw, int dil,
               int ostride, int ocoff, int realC)
{
    using namespace cfg;
    constexpr int NO = NT * 16;
    __shared__ __align__(16) __bf16 sm[2 * NO * 32];

    const int tid   = threadIdx.x;
    const int lane  = tid & 31;
    const int wv    = tid >> 5;
    const int mtile = blockIdx.x * 8 + wv;
    const int n0g   = blockIdx.y * NO;
    const int cpt   = Cin >> 5;           // 32-wide K chunks per tap
    const int NC    = khw * cpt;

    // A-side lane geometry (row M = lane & 15)
    const int mrow = lane & 15;
    const int p    = mtile * 16 + mrow;
    const int bimg = p / HW;
    const int hw   = p - bimg * HW;
    const int h0   = hw / W;
    const int w0   = hw - h0 * W;
    const int s1   = (lane < 16) ? 0 : 8;   // A sub-chunk select
    const int kb   = (lane < 16) ? 0 : 16;  // B K-half select

    // LDS staging geometry: thread -> one 16B async copy per chunk
    const int so = tid >> 2;
    const int sk = (tid & 3) * 8;

    auto stage = [&](int kc, int nb) {
        if (so < NO) {
            int t  = kc / cpt;
            int c0 = (kc - t * cpt) << 5;
            const __bf16* g = wp + ((size_t)(t * CoutP + n0g + so)) * Cin + c0 + sk;
            unsigned la = (unsigned)(uintptr_t)(sm + nb * (NO * 32) + so * 32 + sk);
            asm volatile("global_load_async_to_lds_b128 %0, %1, off"
                         :: "v"(la), "v"((const void*)g) : "memory");
        }
    };

    v8f acc[NT];
#pragma unroll
    for (int j = 0; j < NT; ++j) {
        float bv = bias ? bias[n0g + j * 16 + mrow] : 0.f;
        v8f t = {bv, bv, bv, bv, bv, bv, bv, bv};
        acc[j] = t;
    }

    stage(0, 0);
    for (int kc = 0; kc < NC; ++kc) {
        const int nb = kc & 1;
        if (kc + 1 < NC) {
            stage(kc + 1, nb ^ 1);
            asm volatile("s_wait_asynccnt 1" ::: "memory");
        } else {
            asm volatile("s_wait_asynccnt 0" ::: "memory");
        }
        __syncthreads();

        // ---- A fragment (zero-padded 3x3/dilated tap) ----
        const int t  = kc / cpt;
        const int c0 = (kc - t * cpt) << 5;
        int ky = 0, kx = 0;
        if (khw == 9) { ky = t / 3 - 1; kx = t - (t / 3) * 3 - 1; }
        const int hh = h0 + ky * dil;
        const int ww = w0 + kx * dil;
        v8bf alo = {(__bf16)0.f,(__bf16)0.f,(__bf16)0.f,(__bf16)0.f,
                    (__bf16)0.f,(__bf16)0.f,(__bf16)0.f,(__bf16)0.f};
        v8bf ahi = alo;
        if ((unsigned)hh < (unsigned)H && (unsigned)ww < (unsigned)W) {
            const __bf16* ab = in + (((size_t)bimg * H + hh) * W + ww) * Cin + c0 + s1;
            alo = *(const v8bf*)ab;
            ahi = *(const v8bf*)(ab + 16);
        }
        v16bf afrag = CAT16(alo, ahi);

        // ---- B fragments from LDS, NT tiles reuse A ----
#pragma unroll
        for (int j = 0; j < NT; ++j) {
            const __bf16* sp = sm + nb * (NO * 32) + (j * 16 + mrow) * 32 + s1;
            v8bf blo = *(const v8bf*)sp;
            v8bf bhi = *(const v8bf*)(sp + 16);
            v16bf bfrag = CAT16(blo, bhi);
            acc[j] = __builtin_amdgcn_wmma_f32_16x16x32_bf16(
                         false, afrag, false, bfrag, (short)0, acc[j], false, false);
        }
        __syncthreads();
    }

    // ---- epilogue: C/D layout lane(n = lane&15), vgpr r -> m = r + 8*(lane>=16)
    const int hi8 = (lane >> 4) << 3;
#pragma unroll
    for (int j = 0; j < NT; ++j) {
        const int c = n0g + j * 16 + mrow;
#pragma unroll
        for (int r = 0; r < 8; ++r) {
            const int pm = mtile * 16 + r + hi8;
            float v = acc[j][r];
            if (HASRES) v += (float)res[(size_t)pm * ostride + ocoff + c];
            if (RELU)   v = v > 0.f ? v : 0.f;
            if (OUTMODE == 0) {
                ((__bf16*)outv)[(size_t)pm * ostride + ocoff + c] = (__bf16)v;
            } else if (OUTMODE == 1) {
                ((float*)outv)[(size_t)pm * ostride + c] = v;
            } else {
                if (c < realC) {
                    int bi = pm / HW, phw = pm - bi * HW;
                    ((float*)outv)[((size_t)bi * realC + c) * HW + phw] = v;
                }
            }
        }
    }
}

// ===================== deformable grouped conv (WMMA) =======================
// Per wave: 16 positions x 64 outputs (one group). K = 9 taps * 64 ch = 576.
__global__ __launch_bounds__(256)
void dcn_wmma(const __bf16* __restrict__ in, const float* __restrict__ off,
              const __bf16* __restrict__ wp, __bf16* __restrict__ out)
{
    using namespace cfg;
    const int tid = threadIdx.x, lane = tid & 31, wv = tid >> 5;
    const int mtile = blockIdx.x * 8 + wv;
    const int g = blockIdx.y;                 // group 0..3
    const int mrow = lane & 15;
    const int p = mtile * 16 + mrow;
    const int bimg = p / HW;
    const int hw = p - bimg * HW;
    const int h0 = hw / W, w0 = hw - h0 * W;
    const int s1 = (lane < 16) ? 0 : 8;
    const int kb = (lane < 16) ? 0 : 16;

    v8f z = {0.f,0.f,0.f,0.f,0.f,0.f,0.f,0.f};
    v8f acc[4] = {z, z, z, z};

    for (int kk = 0; kk < 9; ++kk) {
        const float dy = off[(size_t)p * 32 + 2 * kk];
        const float dx = off[(size_t)p * 32 + 2 * kk + 1];
        const float py = (float)(h0 + kk / 3 - 1) + dy;
        const float px = (float)(w0 + kk % 3 - 1) + dx;
        const float y0f = floorf(py), x0f = floorf(px);
        const int y0 = (int)y0f, x0 = (int)x0f;
        const float fy = py - y0f, fx = px - x0f;
        const float w00 = (1.f - fy) * (1.f - fx), w01 = (1.f - fy) * fx;
        const float w10 = fy * (1.f - fx),         w11 = fy * fx;

        bool v00, v01, v10, v11;
        auto corner = [&](int y, int x, bool& vld) -> const __bf16* {
            vld = ((unsigned)y < (unsigned)H) && ((unsigned)x < (unsigned)W);
            int yc = y < 0 ? 0 : (y > H - 1 ? H - 1 : y);
            int xc = x < 0 ? 0 : (x > W - 1 ? W - 1 : x);
            return in + (((size_t)bimg * H + yc) * W + xc) * 256;
        };
        const __bf16* p00 = corner(y0,     x0,     v00);
        const __bf16* p01 = corner(y0,     x0 + 1, v01);
        const __bf16* p10 = corner(y0 + 1, x0,     v10);
        const __bf16* p11 = corner(y0 + 1, x0 + 1, v11);

#pragma unroll
        for (int half = 0; half < 2; ++half) {
            const int cs = half * 32;
            auto samp = [&](int co) -> v8bf {
                v8f r = {0.f,0.f,0.f,0.f,0.f,0.f,0.f,0.f};
                if (v00) r += w00 * __builtin_convertvector(*(const v8bf*)(p00 + co), v8f);
                if (v01) r += w01 * __builtin_convertvector(*(const v8bf*)(p01 + co), v8f);
                if (v10) r += w10 * __builtin_convertvector(*(const v8bf*)(p10 + co), v8f);
                if (v11) r += w11 * __builtin_convertvector(*(const v8bf*)(p11 + co), v8f);
                return __builtin_convertvector(r, v8bf);
            };
            const int cb = g * 64 + cs + s1;
            v16bf afrag = CAT16(samp(cb), samp(cb + 16));
#pragma unroll
            for (int j = 0; j < 4; ++j) {
                const __bf16* bp = wp + ((size_t)(g * 64 + j * 16 + mrow)) * 576
                                      + kk * 64 + cs + s1;
                v16bf bfrag = CAT16(*(const v8bf*)bp, *(const v8bf*)(bp + 16));
                acc[j] = __builtin_amdgcn_wmma_f32_16x16x32_bf16(
                             false, afrag, false, bfrag, (short)0, acc[j], false, false);
            }
        }
    }

    const int hi8 = (lane >> 4) << 3;
#pragma unroll
    for (int j = 0; j < 4; ++j)
#pragma unroll
        for (int r = 0; r < 8; ++r)
            out[(size_t)(mtile * 16 + r + hi8) * 256 + g * 64 + j * 16 + mrow]
                = (__bf16)acc[j][r];
}

// ============================ helper kernels ================================
__global__ void cvt_in(const float* __restrict__ x, __bf16* __restrict__ o)
{   // NCHW f32 [8,512,32,88] -> NHWC bf16 [P,512]
    using namespace cfg;
    size_t idx = (size_t)blockIdx.x * blockDim.x + threadIdx.x;
    if (idx >= (size_t)P * 512) return;
    int pp = (int)(idx >> 9), c = (int)(idx & 511);
    int b = pp / HW, phw = pp - b * HW;
    o[idx] = (__bf16)x[((size_t)b * 512 + c) * HW + phw];
}

__global__ void repack_w(const float* __restrict__ w, const float* __restrict__ scale,
                         __bf16* __restrict__ wp, int O, int OP, int Cin, int khw)
{   // OIHW f32 -> [tap][OP][Cin] bf16 (scale folded, zero-padded rows)
    size_t total = (size_t)khw * OP * Cin;
    size_t idx = (size_t)blockIdx.x * blockDim.x + threadIdx.x;
    if (idx >= total) return;
    int t = (int)(idx / ((size_t)OP * Cin));
    int r = (int)(idx - (size_t)t * OP * Cin);
    int o = r / Cin, i = r - o * Cin;
    float v = 0.f;
    if (o < O) {
        v = w[((size_t)o * Cin + i) * khw + t];
        if (scale) v *= scale[o];
    }
    wp[idx] = (__bf16)v;
}

__global__ void repack_wdcn(const float* __restrict__ w, __bf16* __restrict__ wp)
{   // [256][64][3][3] -> [o_global][kk*64 + c]
    size_t idx = (size_t)blockIdx.x * blockDim.x + threadIdx.x;
    if (idx >= (size_t)256 * 576) return;
    int o = (int)(idx / 576), r = (int)(idx - (size_t)o * 576);
    int kk = r / 64, c = r - kk * 64;
    wp[idx] = (__bf16)w[((size_t)o * 64 + c) * 9 + kk];
}

__global__ void fold_bias(const float* cb, const float* sc, const float* bb,
                          float* o, int nreal, int npad)
{
    int i = threadIdx.x;
    if (i >= npad) return;
    float v = 0.f;
    if (i < nreal) {
        float base = cb ? cb[i] : 0.f;
        if (sc) base *= sc[i];
        v = base + (bb ? bb[i] : 0.f);
    }
    o[i] = v;
}

__global__ void pool_mean(const __bf16* __restrict__ h, float* __restrict__ gp)
{
    using namespace cfg;
    int idx = blockIdx.x * blockDim.x + threadIdx.x;
    if (idx >= B * 256) return;
    int b = idx >> 8, c = idx & 255;
    const __bf16* base = h + (size_t)b * HW * 256 + c;
    float s = 0.f;
    for (int i = 0; i < HW; ++i) s += (float)base[(size_t)i * 256];
    gp[idx] = s * (1.f / HW);
}

__global__ void pool_fc(const float* __restrict__ gp, const float* __restrict__ apw,
                        const float* __restrict__ sc, const float* __restrict__ bb,
                        __bf16* __restrict__ b5)
{
    using namespace cfg;
    int idx = blockIdx.x * blockDim.x + threadIdx.x;
    if (idx >= B * 256) return;
    int b = idx >> 8, o = idx & 255;
    float a = 0.f;
    for (int c = 0; c < 256; ++c) a += gp[b * 256 + c] * apw[(size_t)o * 256 + c];
    a = a * sc[o] + bb[o];
    a = a > 0.f ? a : 0.f;
    b5[idx] = (__bf16)a;
}

__global__ void bcast_b5(const __bf16* __restrict__ b5, __bf16* __restrict__ cat)
{
    using namespace cfg;
    size_t idx = (size_t)blockIdx.x * blockDim.x + threadIdx.x;
    if (idx >= (size_t)P * 256) return;
    int pp = (int)(idx >> 8), c = (int)(idx & 255);
    int b = pp / HW;
    cat[(size_t)pp * 1280 + 1024 + c] = b5[b * 256 + c];
}

// ============================== launch ======================================
extern "C" void kernel_launch(void* const* d_in, const int* in_sizes, int n_in,
                              void* d_out, int out_size, void* d_ws, size_t ws_size,
                              hipStream_t stream)
{
    using namespace cfg;
    (void)in_sizes; (void)n_in; (void)out_size; (void)ws_size;

    // ---- inputs (setup_inputs dict order) ----
    const float* x     = (const float*)d_in[0];
    const float* rc_w  = (const float*)d_in[1];
    const float* rc_b  = (const float*)d_in[2];
    const float* rc_s  = (const float*)d_in[3];
    const float* rc_sb = (const float*)d_in[4];
    // bb i: base 5 + (i-1)*6 : w1,s1,b1,w2,s2,b2
    const float* a1_w = (const float*)d_in[23];
    const float* a1_s = (const float*)d_in[24];
    const float* a1_b = (const float*)d_in[25];
    const float* aw[3]  = {(const float*)d_in[26], (const float*)d_in[29], (const float*)d_in[32]};
    const float* as_[3] = {(const float*)d_in[27], (const float*)d_in[30], (const float*)d_in[33]};
    const float* ab_[3] = {(const float*)d_in[28], (const float*)d_in[31], (const float*)d_in[34]};
    const float* ap_w    = (const float*)d_in[35];
    const float* ap_s    = (const float*)d_in[36];
    const float* ap_b    = (const float*)d_in[37];
    const float* aprj_w  = (const float*)d_in[38];
    const float* aprj_s  = (const float*)d_in[39];
    const float* aprj_b  = (const float*)d_in[40];
    const float* off_w   = (const float*)d_in[41];
    const float* off_b   = (const float*)d_in[42];
    const float* dcn_w   = (const float*)d_in[43];
    const float* fc_w    = (const float*)d_in[44];
    const float* fc_b    = (const float*)d_in[45];

    // ---- workspace layout (bytes) ----
    char* ws = (char*)d_ws;
    constexpr size_t SZ_INX = (size_t)P * 512 * 2;
    constexpr size_t SZ_ACT = (size_t)P * 256 * 2;
    constexpr size_t SZ_CAT = (size_t)P * 1280 * 2;
    constexpr size_t SZ_OFF = (size_t)P * 32 * 4;
    constexpr size_t O_INX = 0;
    constexpr size_t O_HA  = O_INX + SZ_INX;
    constexpr size_t O_HT  = O_HA + SZ_ACT;
    constexpr size_t O_CAT = O_HT + SZ_ACT;
    constexpr size_t O_OFB = O_CAT + SZ_CAT;
    constexpr size_t O_DCO = O_OFB + SZ_OFF;
    constexpr size_t O_WRC = O_DCO + SZ_ACT;
    constexpr size_t O_WBB = O_WRC + (size_t)9 * 256 * 512 * 2;
    constexpr size_t SZ_W33 = (size_t)9 * 256 * 256 * 2;
    constexpr size_t O_WA1 = O_WBB + 6 * SZ_W33;
    constexpr size_t O_WA2 = O_WA1 + (size_t)256 * 256 * 2;
    constexpr size_t O_WPR = O_WA2 + 3 * SZ_W33;
    constexpr size_t O_WOF = O_WPR + (size_t)1280 * 256 * 2;
    constexpr size_t O_WDC = O_WOF + (size_t)9 * 32 * 256 * 2;
    constexpr size_t O_WFC = O_WDC + (size_t)256 * 576 * 2;
    constexpr size_t O_BIA = O_WFC + (size_t)128 * 256 * 2;
    constexpr size_t O_GP  = O_BIA + 14 * 1024;
    constexpr size_t O_B5  = O_GP + (size_t)8 * 256 * 4;

    __bf16* inX  = (__bf16*)(ws + O_INX);
    __bf16* hA   = (__bf16*)(ws + O_HA);
    __bf16* hT   = (__bf16*)(ws + O_HT);
    __bf16* cat  = (__bf16*)(ws + O_CAT);
    float*  ofb  = (float*)(ws + O_OFB);
    __bf16* dco  = (__bf16*)(ws + O_DCO);
    __bf16* wpRC = (__bf16*)(ws + O_WRC);
    __bf16* wpA1 = (__bf16*)(ws + O_WA1);
    __bf16* wpPR = (__bf16*)(ws + O_WPR);
    __bf16* wpOF = (__bf16*)(ws + O_WOF);
    __bf16* wpDC = (__bf16*)(ws + O_WDC);
    __bf16* wpFC = (__bf16*)(ws + O_WFC);
    float*  gp   = (float*)(ws + O_GP);
    __bf16* b5   = (__bf16*)(ws + O_B5);
    auto biasSlot = [&](int i) -> float* { return (float*)(ws + O_BIA + (size_t)i * 1024); };

    const dim3 blk(256);
    auto nb = [](size_t n) { return (unsigned)((n + 255) / 256); };

    // ---- input + weight repacks (all on stream) ----
    cvt_in<<<nb((size_t)P * 512), blk, 0, stream>>>(x, inX);

    repack_w<<<nb((size_t)9 * 256 * 512), blk, 0, stream>>>(rc_w, rc_s, wpRC, 256, 256, 512, 9);
    fold_bias<<<1, blk, 0, stream>>>(rc_b, rc_s, rc_sb, biasSlot(0), 256, 256);

    for (int i = 0; i < 3; ++i) {
        const float* w1 = (const float*)d_in[5 + i * 6 + 0];
        const float* s1 = (const float*)d_in[5 + i * 6 + 1];
        const float* b1 = (const float*)d_in[5 + i * 6 + 2];
        const float* w2 = (const float*)d_in[5 + i * 6 + 3];
        const float* s2 = (const float*)d_in[5 + i * 6 + 4];
        const float* b2 = (const float*)d_in[5 + i * 6 + 5];
        __bf16* wp1 = (__bf16*)(ws + O_WBB + (size_t)(i * 2 + 0) * SZ_W33);
        __bf16* wp2 = (__bf16*)(ws + O_WBB + (size_t)(i * 2 + 1) * SZ_W33);
        repack_w<<<nb((size_t)9 * 256 * 256), blk, 0, stream>>>(w1, s1, wp1, 256, 256, 256, 9);
        repack_w<<<nb((size_t)9 * 256 * 256), blk, 0, stream>>>(w2, s2, wp2, 256, 256, 256, 9);
        fold_bias<<<1, blk, 0, stream>>>(nullptr, nullptr, b1, biasSlot(1 + i * 2), 256, 256);
        fold_bias<<<1, blk, 0, stream>>>(nullptr, nullptr, b2, biasSlot(2 + i * 2), 256, 256);
    }
    repack_w<<<nb((size_t)256 * 256), blk, 0, stream>>>(a1_w, a1_s, wpA1, 256, 256, 256, 1);
    fold_bias<<<1, blk, 0, stream>>>(nullptr, nullptr, a1_b, biasSlot(7), 256, 256);
    for (int j = 0; j < 3; ++j) {
        __bf16* wpa = (__bf16*)(ws + O_WA2 + (size_t)j * SZ_W33);
        repack_w<<<nb((size_t)9 * 256 * 256), blk, 0, stream>>>(aw[j], as_[j], wpa, 256, 256, 256, 9);
        fold_bias<<<1, blk, 0, stream>>>(nullptr, nullptr, ab_[j], biasSlot(8 + j), 256, 256);
    }
    repack_w<<<nb((size_t)256 * 1280), blk, 0, stream>>>(aprj_w, aprj_s, wpPR, 256, 256, 1280, 1);
    fold_bias<<<1, blk, 0, stream>>>(nullptr, nullptr, aprj_b, biasSlot(11), 256, 256);
    repack_w<<<nb((size_t)9 * 32 * 256), blk, 0, stream>>>(off_w, nullptr, wpOF, 18, 32, 256, 9);
    fold_bias<<<1, blk, 0, stream>>>(off_b, nullptr, nullptr, biasSlot(12), 18, 32);
    repack_wdcn<<<nb((size_t)256 * 576), blk, 0, stream>>>(dcn_w, wpDC);
    repack_w<<<nb((size_t)128 * 256), blk, 0, stream>>>(fc_w, nullptr, wpFC, 118, 128, 256, 1);
    fold_bias<<<1, blk, 0, stream>>>(fc_b, nullptr, nullptr, biasSlot(13), 118, 128);

    // ---- network ----
    // reduce_conv: 3x3 512->256 + BN + ReLU
    conv_wmma<4, 0, true, false><<<dim3(MBLK, 4), blk, 0, stream>>>(
        inX, wpRC, biasSlot(0), nullptr, hA, 512, 256, 9, 1, 256, 0, 256);

    // 3 BasicBlocks (conv2 adds residual pre-ReLU, writes hA in place)
    for (int i = 0; i < 3; ++i) {
        __bf16* wp1 = (__bf16*)(ws + O_WBB + (size_t)(i * 2 + 0) * SZ_W33);
        __bf16* wp2 = (__bf16*)(ws + O_WBB + (size_t)(i * 2 + 1) * SZ_W33);
        conv_wmma<4, 0, true, false><<<dim3(MBLK, 4), blk, 0, stream>>>(
            hA, wp1, biasSlot(1 + i * 2), nullptr, hT, 256, 256, 9, 1, 256, 0, 256);
        conv_wmma<4, 0, true, true><<<dim3(MBLK, 4), blk, 0, stream>>>(
            hT, wp2, biasSlot(2 + i * 2), hA, hA, 256, 256, 9, 1, 256, 0, 256);
    }

    // ASPP branches -> cat[P][1280]
    conv_wmma<4, 0, true, false><<<dim3(MBLK, 4), blk, 0, stream>>>(
        hA, wpA1, biasSlot(7), nullptr, cat, 256, 256, 1, 1, 1280, 0, 256);
    const int dils[3] = {6, 12, 18};
    for (int j = 0; j < 3; ++j) {
        __bf16* wpa = (__bf16*)(ws + O_WA2 + (size_t)j * SZ_W33);
        conv_wmma<4, 0, true, false><<<dim3(MBLK, 4), blk, 0, stream>>>(
            hA, wpa, biasSlot(8 + j), nullptr, cat, 256, 256, 9, dils[j],
            1280, 256 * (j + 1), 256);
    }
    pool_mean<<<nb(B * 256), blk, 0, stream>>>(hA, gp);
    pool_fc<<<nb(B * 256), blk, 0, stream>>>(gp, ap_w, ap_s, ap_b, b5);
    bcast_b5<<<nb((size_t)P * 256), blk, 0, stream>>>(b5, cat);

    // ASPP projection 1x1 1280->256 -> hT
    conv_wmma<4, 0, true, false><<<dim3(MBLK, 4), blk, 0, stream>>>(
        cat, wpPR, biasSlot(11), nullptr, hT, 1280, 256, 1, 1, 256, 0, 256);

    // offset conv 3x3 256->18 (padded 32), f32 NHWC stride 32
    conv_wmma<2, 1, false, false><<<dim3(MBLK, 1), blk, 0, stream>>>(
        hT, wpOF, biasSlot(12), nullptr, ofb, 256, 32, 9, 1, 32, 0, 18);

    // deformable grouped conv -> dco
    dcn_wmma<<<dim3(MBLK, 4), blk, 0, stream>>>(hT, ofb, wpDC, dco);

    // final 1x1 256->118, f32 NCHW into d_out
    conv_wmma<4, 2, false, false><<<dim3(MBLK, 2), blk, 0, stream>>>(
        dco, wpFC, biasSlot(13), nullptr, (float*)d_out, 256, 128, 1, 1, 0, 0, 118);
}
